// CCM_74663711474120
// MI455X (gfx1250) — matrix operand order repaired and checked by reading, not compiled, not run on hardware
//
#include <hip/hip_runtime.h>
#include <hip/hip_bf16.h>
#include <math.h>

typedef _Float16 v16h __attribute__((ext_vector_type(16)));
typedef _Float16 v8h  __attribute__((ext_vector_type(8)));
typedef float    v8f  __attribute__((ext_vector_type(8)));
typedef unsigned int u32x4 __attribute__((ext_vector_type(4)));
typedef unsigned int u32x8 __attribute__((ext_vector_type(8)));

union V16H { v16h v; v8h h8[2]; _Float16 h[16]; };
union V8F  { v8f v; float f[8]; };

constexpr int DD  = 2048;   // feature dim
constexpr int NN  = 1024;   // memory queue length
constexpr int BSZ = 64;     // batch
constexpr int CC  = 65;     // classes
constexpr int HH  = 128;    // hidden
constexpr int CP  = 80;     // classes padded to 5*16 for WMMA col tiles
#define BN_EPS 1e-5f

// ---------------------------------------------------------------------------
// f32 -> f16 conversions + head_w1 transpose
// ---------------------------------------------------------------------------
__global__ void k_convert(const float* __restrict__ emb, const float* __restrict__ allf,
                          const float* __restrict__ w1,
                          _Float16* __restrict__ emb_h, _Float16* __restrict__ allf_h,
                          _Float16* __restrict__ w1T_h) {
  long i = (long)blockIdx.x * blockDim.x + threadIdx.x;
  const long nE = (long)NN * DD, nF = (long)BSZ * DD, nW = (long)HH * DD;
  if (i < nE) {
    emb_h[i] = (_Float16)emb[i];
  } else if (i < nE + nF) {
    long j = i - nE;
    allf_h[j] = (_Float16)allf[j];
  } else if (i < nE + nF + nW) {
    long j = i - nE - nF;
    int n = (int)(j / DD), k = (int)(j % DD);
    w1T_h[j] = (_Float16)w1[(long)k * HH + n];   // [128][2048] = head_w1^T
  }
}

// ---------------------------------------------------------------------------
// per-row inverse L2 norms (rows 0..63 = all_f, 64..1087 = embedding)
// ---------------------------------------------------------------------------
__global__ void k_rownorm(const float* __restrict__ allf, const float* __restrict__ emb,
                          float* __restrict__ inv_fn, float* __restrict__ inv_en) {
  int row = blockIdx.x, l = threadIdx.x;
  const float* src = (row < BSZ) ? allf + (long)row * DD : emb + (long)(row - BSZ) * DD;
  float s = 0.f;
  for (int k = l; k < DD; k += 32) { float v = src[k]; s += v * v; }
  for (int m = 16; m >= 1; m >>= 1) s += __shfl_xor(s, m, 32);
  if (l == 0) {
    float nrm = fmaxf(sqrtf(s), 1e-12f);
    float inv = 1.f / nrm;
    if (row < BSZ) inv_fn[row] = inv; else inv_en[row - BSZ] = inv;
  }
}

// ---------------------------------------------------------------------------
// fold BN affine params + small bias-vector matmuls (1 block, 128 threads)
// ---------------------------------------------------------------------------
__global__ void k_small(const float* g, const float* beta, const float* rm, const float* rv,
                        const float* b1,
                        const float* mg, const float* mbeta, const float* mrm, const float* mrv,
                        const float* b2, const float* mw1, const float* mb1,
                        float* svec, float* uvec, float* s2vec, float* t2vec,
                        float* bz, float* bfv) {
  int j = threadIdx.x;
  float s = g[j] * rsqrtf(rv[j] + BN_EPS);
  svec[j] = s;
  uvec[j] = b1[j] * s + beta[j] - rm[j] * s;
  float s2 = mg[j] * rsqrtf(mrv[j] + BN_EPS);
  s2vec[j] = s2;
  t2vec[j] = mbeta[j] - mrm[j] * s2;
  float az = 0.f, af = 0.f;
  for (int i = 0; i < HH; i++) {
    float b = b2[i];
    az += b * mw1[(HH + i) * HH + j];   // head_b2 @ mix_w1[H:]
    af += b * mw1[i * HH + j];          // head_b2 @ mix_w1[:H]
  }
  bz[j] = az;
  bfv[j] = mb1[j] + af;
}

// ---------------------------------------------------------------------------
// W2M^T (f16, n-major) = (head_w2 @ mix_w1[H:])^T ; W2Mf (f32) = head_w2 @ mix_w1[:H]
// ---------------------------------------------------------------------------
__global__ void k_w2m(const float* __restrict__ w2, const float* __restrict__ mw1,
                      _Float16* __restrict__ W2MT_h, float* __restrict__ W2Mf) {
  int t = blockIdx.x * blockDim.x + threadIdx.x;
  if (t < HH * HH) {
    int n = t & (HH - 1), k = t >> 7;
    float a = 0.f;
    for (int j = 0; j < HH; j++) a += w2[k * HH + j] * mw1[(HH + j) * HH + n];
    W2MT_h[n * HH + k] = (_Float16)a;
  } else if (t < 2 * HH * HH) {
    int tt = t - HH * HH;
    int j2 = tt & (HH - 1), j = tt >> 7;
    float a = 0.f;
    for (int i = 0; i < HH; i++) a += w2[j * HH + i] * mw1[i * HH + j2];
    W2Mf[j * HH + j2] = a;
  }
}

// ---------------------------------------------------------------------------
// W2C^T (f16, c-major, zero-padded to 80 cols) = (mix_w2 @ clf_w)^T ; c_vec
// ---------------------------------------------------------------------------
__global__ void k_w2c(const float* __restrict__ mw2, const float* __restrict__ clfw,
                      const float* __restrict__ mb2, const float* __restrict__ clfb,
                      _Float16* __restrict__ W2CT_h, float* __restrict__ cvec) {
  int t = blockIdx.x * blockDim.x + threadIdx.x;
  if (t < CP * HH) {
    int c = t / HH, k = t % HH;
    float a = 0.f;
    if (c < CC)
      for (int d = 0; d < DD; d++) a += mw2[(long)k * DD + d] * clfw[(long)d * CC + c];
    W2CT_h[c * HH + k] = (_Float16)a;
  } else if (t < CP * HH + CP) {
    int c = t - CP * HH;
    float a = 0.f;
    if (c < CC) {
      a = clfb[c];
      for (int d = 0; d < DD; d++) a += mb2[d] * clfw[(long)d * CC + c];
    }
    cvec[c] = a;
  }
}

// ---------------------------------------------------------------------------
// WMMA: Z[b,n] = <f_b, e_n> * inv_fn[b] * inv_en[n] / (T*sqrt(N))   [64 x 1024]
// 16x64 output tile per wave; all 4 B fragments prefetched into distinct regs
// so wmma t only waits on its own pair of loads (in-order load completion).
// grid: (16 n-groups, 4 b-tiles), 1 wave each
// ---------------------------------------------------------------------------
__global__ void __launch_bounds__(32)
k_attn(const _Float16* __restrict__ allf_h, const _Float16* __restrict__ emb_h,
       const float* __restrict__ inv_fn, const float* __restrict__ inv_en,
       float* __restrict__ zbuf) {
  int ng = blockIdx.x, bt = blockIdx.y, l = threadIdx.x;
  int hi = l >> 4, lo = l & 15;
  const _Float16* ap  = allf_h + (long)(bt * 16 + lo) * DD + hi * 8;    // A row = b
  const _Float16* bp0 = emb_h + (long)(ng * 64 + lo) * DD + hi * 8;     // B col = n
  V8F acc[4];
#pragma unroll
  for (int t = 0; t < 4; t++)
#pragma unroll
    for (int i = 0; i < 8; i++) acc[t].f[i] = 0.f;
  for (int kk = 0; kk < DD; kk += 32) {
    V16H a, b[4];
    a.h8[0] = *(const v8h*)(ap + kk);
    a.h8[1] = *(const v8h*)(ap + kk + 16);
#pragma unroll
    for (int t = 0; t < 4; t++) {
      const _Float16* bp = bp0 + (long)t * 16 * DD;
      b[t].h8[0] = *(const v8h*)(bp + kk);
      b[t].h8[1] = *(const v8h*)(bp + kk + 16);
    }
#pragma unroll
    for (int t = 0; t < 4; t++)
      acc[t].v = __builtin_amdgcn_wmma_f32_16x16x32_f16(false, a.v, false, b[t].v,
                                                        (short)0, acc[t].v, false, false);
  }
  const float invTS = 1.f / (0.07f * 32.f);
#pragma unroll
  for (int t = 0; t < 4; t++) {
    int n = ng * 64 + t * 16 + lo;
    float ien = inv_en[n] * invTS;
#pragma unroll
    for (int j = 0; j < 8; j++) {
      int b_ = bt * 16 + hi * 8 + j;
      zbuf[(long)b_ * NN + n] = acc[t].f[j] * inv_fn[b_] * ien;
    }
  }
}

// ---------------------------------------------------------------------------
// row softmax over N=1024 (in place); one 256-thread block per row
// ---------------------------------------------------------------------------
__global__ void k_softmax(float* __restrict__ zbuf) {
  __shared__ float red[256];
  int row = blockIdx.x, t = threadIdx.x;
  float* p = zbuf + (long)row * NN;
  float mx = -1e30f;
  for (int n = t; n < NN; n += 256) mx = fmaxf(mx, p[n]);
  red[t] = mx; __syncthreads();
  for (int s = 128; s > 0; s >>= 1) { if (t < s) red[t] = fmaxf(red[t], red[t + s]); __syncthreads(); }
  mx = red[0]; __syncthreads();
  float sm = 0.f;
  for (int n = t; n < NN; n += 256) { float e = __expf(p[n] - mx); p[n] = e; sm += e; }
  red[t] = sm; __syncthreads();
  for (int s = 128; s > 0; s >>= 1) { if (t < s) red[t] += red[t + s]; __syncthreads(); }
  float inv = 1.f / red[0];
  for (int n = t; n < NN; n += 256) p[n] *= inv;
}

// ---------------------------------------------------------------------------
// WMMA: Y = emb @ head_w1 -> fused BN affine + relu -> f16 act [1024 x 128]
// 16x64 tile per wave; grid: (2 n-groups, 64 m-tiles)
// ---------------------------------------------------------------------------
__global__ void __launch_bounds__(32)
k_heady(const _Float16* __restrict__ emb_h, const _Float16* __restrict__ w1T_h,
        const float* __restrict__ svec, const float* __restrict__ uvec,
        _Float16* __restrict__ act_h) {
  int ng = blockIdx.x, mt = blockIdx.y, l = threadIdx.x;
  int hi = l >> 4, lo = l & 15;
  const _Float16* ap  = emb_h + (long)(mt * 16 + lo) * DD + hi * 8;
  const _Float16* bp0 = w1T_h + (long)(ng * 64 + lo) * DD + hi * 8;
  V8F acc[4];
#pragma unroll
  for (int t = 0; t < 4; t++)
#pragma unroll
    for (int i = 0; i < 8; i++) acc[t].f[i] = 0.f;
  for (int kk = 0; kk < DD; kk += 32) {
    V16H a, b[4];
    a.h8[0] = *(const v8h*)(ap + kk);
    a.h8[1] = *(const v8h*)(ap + kk + 16);
#pragma unroll
    for (int t = 0; t < 4; t++) {
      const _Float16* bp = bp0 + (long)t * 16 * DD;
      b[t].h8[0] = *(const v8h*)(bp + kk);
      b[t].h8[1] = *(const v8h*)(bp + kk + 16);
    }
#pragma unroll
    for (int t = 0; t < 4; t++)
      acc[t].v = __builtin_amdgcn_wmma_f32_16x16x32_f16(false, a.v, false, b[t].v,
                                                        (short)0, acc[t].v, false, false);
  }
#pragma unroll
  for (int t = 0; t < 4; t++) {
    int col = ng * 64 + t * 16 + lo;
    float s = svec[col], u = uvec[col];
#pragma unroll
    for (int j = 0; j < 8; j++) {
      int m = mt * 16 + hi * 8 + j;
      act_h[(long)m * HH + col] = (_Float16)fmaxf(acc[t].f[j] * s + u, 0.f);
    }
  }
}

// ---------------------------------------------------------------------------
// WMMA: h1z = act @ W2M + bz  [1024 x 128], K=128; 16x64 tile per wave
// grid: (2 n-groups, 64 m-tiles)
// ---------------------------------------------------------------------------
__global__ void __launch_bounds__(32)
k_h1z(const _Float16* __restrict__ act_h, const _Float16* __restrict__ W2MT_h,
      const float* __restrict__ bz, float* __restrict__ h1z) {
  int ng = blockIdx.x, mt = blockIdx.y, l = threadIdx.x;
  int hi = l >> 4, lo = l & 15;
  const _Float16* ap  = act_h + (long)(mt * 16 + lo) * HH + hi * 8;
  const _Float16* bp0 = W2MT_h + (long)(ng * 64 + lo) * HH + hi * 8;
  V8F acc[4];
#pragma unroll
  for (int t = 0; t < 4; t++)
#pragma unroll
    for (int i = 0; i < 8; i++) acc[t].f[i] = 0.f;
#pragma unroll
  for (int kk = 0; kk < HH; kk += 32) {
    V16H a, b[4];
    a.h8[0] = *(const v8h*)(ap + kk);
    a.h8[1] = *(const v8h*)(ap + kk + 16);
#pragma unroll
    for (int t = 0; t < 4; t++) {
      const _Float16* bp = bp0 + (long)t * 16 * HH;
      b[t].h8[0] = *(const v8h*)(bp + kk);
      b[t].h8[1] = *(const v8h*)(bp + kk + 16);
    }
#pragma unroll
    for (int t = 0; t < 4; t++)
      acc[t].v = __builtin_amdgcn_wmma_f32_16x16x32_f16(false, a.v, false, b[t].v,
                                                        (short)0, acc[t].v, false, false);
  }
#pragma unroll
  for (int t = 0; t < 4; t++) {
    int col = ng * 64 + t * 16 + lo;
    float bb = bz[col];
#pragma unroll
    for (int j = 0; j < 8; j++) {
      int m = mt * 16 + hi * 8 + j;
      h1z[(long)m * HH + col] = acc[t].f[j] + bb;
    }
  }
}

// ---------------------------------------------------------------------------
// h1f = relu(BN(all_f @ head_w1)) @ W2Mf + bfv   [64 x 128] (tiny, VALU)
// ---------------------------------------------------------------------------
__global__ void k_h1f(const float* __restrict__ allf, const float* __restrict__ w1,
                      const float* __restrict__ svec, const float* __restrict__ uvec,
                      const float* __restrict__ W2Mf, const float* __restrict__ bfv,
                      float* __restrict__ h1f) {
  __shared__ float act[HH];
  int b = blockIdx.x, j = threadIdx.x;
  float a = 0.f;
  for (int k = 0; k < DD; k++) a += allf[(long)b * DD + k] * w1[(long)k * HH + j];
  act[j] = fmaxf(a * svec[j] + uvec[j], 0.f);
  __syncthreads();
  float o = bfv[j];
  for (int i = 0; i < HH; i++) o += act[i] * W2Mf[i * HH + j];
  h1f[b * HH + j] = o;
}

// ---------------------------------------------------------------------------
// p_x = softmax over batch of row-sums of attn(all_f, all_f)   (tiny, 1 block)
// ---------------------------------------------------------------------------
__global__ void k_px(const float* __restrict__ allf, const float* __restrict__ inv_fn,
                     float* __restrict__ p_x) {
  __shared__ float r4[256];
  __shared__ float rb[BSZ];
  int t = threadIdx.x;
  int b = t >> 2, q = t & 3;
  float acc = 0.f;
  for (int b2 = q * 16; b2 < q * 16 + 16; b2++) {
    const float* pa = allf + (long)b * DD;
    const float* pb = allf + (long)b2 * DD;
    float dot = 0.f;
    for (int k = 0; k < DD; k++) dot += pa[k] * pb[k];
    acc += dot * inv_fn[b2];
  }
  r4[t] = acc; __syncthreads();
  if (t < BSZ)
    rb[t] = (r4[4 * t] + r4[4 * t + 1] + r4[4 * t + 2] + r4[4 * t + 3]) * inv_fn[t]
            * (1.f / (0.07f * 8.f));
  __syncthreads();
  if (t == 0) {
    float mx = -1e30f;
    for (int i = 0; i < BSZ; i++) mx = fmaxf(mx, rb[i]);
    float sm = 0.f;
    for (int i = 0; i < BSZ; i++) { rb[i] = __expf(rb[i] - mx); sm += rb[i]; }
    float inv = 1.f / sm;
    for (int i = 0; i < BSZ; i++) p_x[i] = rb[i] * inv;
  }
}

__global__ void k_zero(float* p, int n) {
  int i = blockIdx.x * blockDim.x + threadIdx.x;
  if (i < n) p[i] = 0.f;
}

// ---------------------------------------------------------------------------
// Main fused kernel: per (n-tile, b): act = relu(BN(h1f[b]+h1z[n])) in regs,
// logits = act @ W2C + c_vec via WMMA (W2C staged into LDS by the Tensor Data
// Mover), per-row softmax over C=65 via shfl, weight by p_x[b], accumulate;
// atomic add into sum_x.  grid: (64 n-tiles, 8 b-groups of 8), 1 wave each
// ---------------------------------------------------------------------------
__global__ void __launch_bounds__(32)
k_main(const float* __restrict__ h1f, const float* __restrict__ h1z,
       const float* __restrict__ s2vec, const float* __restrict__ t2vec,
       const _Float16* __restrict__ W2CT_h, const float* __restrict__ cvec,
       const float* __restrict__ p_x, float* __restrict__ sumx) {
  __shared__ _Float16 lW[CP * HH];
  __shared__ float ls2[HH], lt2[HH];
  int l = threadIdx.x;

  // --- TDM: DMA the 20KB W2C^T panel global -> LDS (one 1D tile of 2560 x 8B)
  {
    unsigned long long ga = (unsigned long long)W2CT_h;
    unsigned lds = (unsigned)(unsigned long long)(&lW[0]);  // flat addr low bits = LDS addr
    u32x4 g0;
    g0[0] = 1u;                                              // count=1, user mode
    g0[1] = lds;                                             // lds_addr
    g0[2] = (unsigned)ga;                                    // global_addr[31:0]
    g0[3] = (unsigned)((ga >> 32) & 0x1FFFFFFu) | (2u << 30);// global_addr[56:32] | type=2
    u32x8 g1;
    const unsigned ELEMS = (unsigned)(CP * HH * 2 / 8);      // 2560 8-byte elements
    g1[0] = 3u << 16;                                        // data_size = 8B
    g1[1] = (ELEMS & 0xFFFFu) << 16;                         // tensor_dim0[15:0]
    g1[2] = (ELEMS >> 16) | (1u << 16);                      // tensor_dim0[31:16] | tensor_dim1=1
    g1[3] = (ELEMS & 0xFFFFu) << 16;                         // tile_dim0
    g1[4] = 1u;                                              // tile_dim1=1, tile_dim2=0
    g1[5] = ELEMS;                                           // tensor_dim0_stride lo
    g1[6] = 0u;
    g1[7] = 0u;
    u32x4 z4 = {0u, 0u, 0u, 0u};                             // dims >=2 unused (tile_dim2=0)
    asm volatile("tensor_load_to_lds %0, %1, %2, %3"
                 :: "s"(g0), "s"(g1), "s"(z4), "s"(z4) : "memory");
    __builtin_amdgcn_s_wait_tensorcnt(0);
  }
  // belt-and-suspenders staging (rewrites the same bytes; numerics never depend
  // on the D# bit-packing above) + BN affine vectors
  {
    const uint4* src = (const uint4*)W2CT_h;
    uint4* dst = (uint4*)lW;
    for (int i = l; i < CP * HH * 2 / 16; i += 32) dst[i] = src[i];
    for (int i = l; i < HH; i += 32) { ls2[i] = s2vec[i]; lt2[i] = t2vec[i]; }
  }
  __syncthreads();

  int n0 = blockIdx.x * 16;
  int bg = blockIdx.y;
  int hi = l >> 4, lo = l & 15;
  int base0 = hi * 8;
  const float* h1zr = h1z + (long)(n0 + lo) * HH;   // A-fragment row = n

  float cl[5];
#pragma unroll
  for (int ct = 0; ct < 5; ct++) cl[ct] = cvec[ct * 16 + lo];

  V8F racc[5];
#pragma unroll
  for (int ct = 0; ct < 5; ct++)
#pragma unroll
    for (int i = 0; i < 8; i++) racc[ct].f[i] = 0.f;

  for (int b = bg * 8; b < bg * 8 + 8; b++) {
    const float* h1fr = h1f + b * HH;
    // build A fragments (relu(BN(h1f+h1z)) in f16), per ISA A-matrix layout
    V16H a[4];
#pragma unroll
    for (int ks = 0; ks < 4; ks++) {
#pragma unroll
      for (int e = 0; e < 16; e++) {
        int k = ks * 32 + (e < 8 ? base0 + e : 16 + base0 + (e - 8));
        float v = (h1fr[k] + h1zr[k]) * ls2[k] + lt2[k];
        a[ks].h[e] = (_Float16)fmaxf(v, 0.f);
      }
    }
    V8F acc[5];
#pragma unroll
    for (int ct = 0; ct < 5; ct++) {
#pragma unroll
      for (int i = 0; i < 8; i++) acc[ct].f[i] = 0.f;
#pragma unroll
      for (int ks = 0; ks < 4; ks++) {
        V16H bf;
        const _Float16* bp = lW + (ct * 16 + lo) * HH + ks * 32 + hi * 8;
        bf.h8[0] = *(const v8h*)bp;
        bf.h8[1] = *(const v8h*)(bp + 16);
        acc[ct].v = __builtin_amdgcn_wmma_f32_16x16x32_f16(false, a[ks].v, false, bf.v,
                                                           (short)0, acc[ct].v, false, false);
      }
#pragma unroll
      for (int i = 0; i < 8; i++) acc[ct].f[i] += cl[ct];
    }
    float pb = p_x[b];
    bool v4 = (lo == 0);   // col tile 4: only c==64 is real
#pragma unroll
    for (int j = 0; j < 8; j++) {
      float mx = -1e30f;
#pragma unroll
      for (int ct = 0; ct < 4; ct++) mx = fmaxf(mx, acc[ct].f[j]);
      if (v4) mx = fmaxf(mx, acc[4].f[j]);
      for (int m = 8; m >= 1; m >>= 1) mx = fmaxf(mx, __shfl_xor(mx, m, 32));
      float e[5], sm = 0.f;
#pragma unroll
      for (int ct = 0; ct < 4; ct++) { e[ct] = __expf(acc[ct].f[j] - mx); sm += e[ct]; }
      e[4] = v4 ? __expf(acc[4].f[j] - mx) : 0.f;
      sm += e[4];
      for (int m = 8; m >= 1; m >>= 1) sm += __shfl_xor(sm, m, 32);
      float sc = pb / sm;
#pragma unroll
      for (int ct = 0; ct < 5; ct++) racc[ct].f[j] += e[ct] * sc;
    }
  }
#pragma unroll
  for (int ct = 0; ct < 5; ct++) {
    int c = ct * 16 + lo;
    if (c < CC) {
#pragma unroll
      for (int j = 0; j < 8; j++) {
        int n = n0 + hi * 8 + j;   // D-matrix row layout
        atomicAdd(&sumx[(long)n * CC + c], racc[ct].f[j]);
      }
    }
  }
}

// ---------------------------------------------------------------------------
// out = z_x @ sum_x   [64 x 65] (tiny)
// ---------------------------------------------------------------------------
__global__ void k_final(const float* __restrict__ zx, const float* __restrict__ sumx,
                        float* __restrict__ out) {
  int b = blockIdx.x, c = threadIdx.x;
  if (c < CC) {
    float a = 0.f;
    for (int n = 0; n < NN; n++) a += zx[(long)b * NN + n] * sumx[(long)n * CC + c];
    out[b * CC + c] = a;
  }
}

// ---------------------------------------------------------------------------
extern "C" void kernel_launch(void* const* d_in, const int* in_sizes, int n_in,
                              void* d_out, int out_size, void* d_ws, size_t ws_size,
                              hipStream_t stream) {
  (void)in_sizes; (void)n_in; (void)out_size; (void)ws_size;
  const float* allf     = (const float*)d_in[0];
  const float* emb      = (const float*)d_in[1];
  /* d_in[2] = all_y (int32) is unused in this branch */
  const float* head_w1  = (const float*)d_in[3];
  const float* head_b1  = (const float*)d_in[4];
  const float* head_g   = (const float*)d_in[5];
  const float* head_beta= (const float*)d_in[6];
  const float* head_rm  = (const float*)d_in[7];
  const float* head_rv  = (const float*)d_in[8];
  const float* head_w2  = (const float*)d_in[9];
  const float* head_b2  = (const float*)d_in[10];
  const float* mix_w1   = (const float*)d_in[11];
  const float* mix_b1   = (const float*)d_in[12];
  const float* mix_g    = (const float*)d_in[13];
  const float* mix_beta = (const float*)d_in[14];
  const float* mix_rm   = (const float*)d_in[15];
  const float* mix_rv   = (const float*)d_in[16];
  const float* mix_w2   = (const float*)d_in[17];
  const float* mix_b2   = (const float*)d_in[18];
  const float* clf_w    = (const float*)d_in[19];
  const float* clf_b    = (const float*)d_in[20];
  float* out = (float*)d_out;

  char* w = (char*)d_ws;
  auto alloc = [&](size_t bytes) -> char* {
    char* p = w; w += (bytes + 255) & ~(size_t)255; return p;
  };
  _Float16* emb_h   = (_Float16*)alloc((size_t)NN * DD * 2);
  _Float16* allf_h  = (_Float16*)alloc((size_t)BSZ * DD * 2);
  _Float16* w1T_h   = (_Float16*)alloc((size_t)HH * DD * 2);
  _Float16* act_h   = (_Float16*)alloc((size_t)NN * HH * 2);
  _Float16* W2MT_h  = (_Float16*)alloc((size_t)HH * HH * 2);
  _Float16* W2CT_h  = (_Float16*)alloc((size_t)CP * HH * 2);
  float* h1z    = (float*)alloc((size_t)NN * HH * 4);
  float* h1f    = (float*)alloc((size_t)BSZ * HH * 4);
  float* zbuf   = (float*)alloc((size_t)BSZ * NN * 4);
  float* sumx   = (float*)alloc((size_t)NN * CC * 4);
  float* W2Mf   = (float*)alloc((size_t)HH * HH * 4);
  float* inv_fn = (float*)alloc(BSZ * 4);
  float* inv_en = (float*)alloc(NN * 4);
  float* p_x    = (float*)alloc(BSZ * 4);
  float* svec   = (float*)alloc(HH * 4);
  float* uvec   = (float*)alloc(HH * 4);
  float* s2vec  = (float*)alloc(HH * 4);
  float* t2vec  = (float*)alloc(HH * 4);
  float* bz     = (float*)alloc(HH * 4);
  float* bfv    = (float*)alloc(HH * 4);
  float* cvec   = (float*)alloc(CP * 4);

  // --- precompute ---
  {
    long total = (long)NN * DD + (long)BSZ * DD + (long)HH * DD;
    int blocks = (int)((total + 255) / 256);
    k_convert<<<blocks, 256, 0, stream>>>(emb, allf, head_w1, emb_h, allf_h, w1T_h);
  }
  k_rownorm<<<BSZ + NN, 32, 0, stream>>>(allf, emb, inv_fn, inv_en);
  k_small<<<1, HH, 0, stream>>>(head_g, head_beta, head_rm, head_rv, head_b1,
                                mix_g, mix_beta, mix_rm, mix_rv,
                                head_b2, mix_w1, mix_b1,
                                svec, uvec, s2vec, t2vec, bz, bfv);
  k_w2m<<<(2 * HH * HH + 255) / 256, 256, 0, stream>>>(head_w2, mix_w1, W2MT_h, W2Mf);
  k_w2c<<<(CP * HH + CP + 255) / 256, 256, 0, stream>>>(mix_w2, clf_w, mix_b2, clf_b,
                                                        W2CT_h, cvec);
  k_zero<<<(NN * CC + 255) / 256, 256, 0, stream>>>(sumx, NN * CC);

  // --- attention z|x (WMMA, 16x64 tiles) + row softmax ---
  k_attn<<<dim3(NN / 64, BSZ / 16), 32, 0, stream>>>(allf_h, emb_h, inv_fn, inv_en, zbuf);
  k_softmax<<<BSZ, 256, 0, stream>>>(zbuf);

  // --- z path: head(embedding) folded through mixer (WMMA, 16x64 tiles) ---
  k_heady<<<dim3(HH / 64, NN / 16), 32, 0, stream>>>(emb_h, w1T_h, svec, uvec, act_h);
  k_h1z<<<dim3(HH / 64, NN / 16), 32, 0, stream>>>(act_h, W2MT_h, bz, h1z);

  // --- f path + p(x) (tiny) ---
  k_h1f<<<BSZ, HH, 0, stream>>>(allf, head_w1, svec, uvec, W2Mf, bfv, h1f);
  k_px<<<1, 256, 0, stream>>>(allf, inv_fn, p_x);

  // --- fused mixer-BN-relu @ W2C + softmax + p_x-weighted accumulate (WMMA+TDM) ---
  k_main<<<dim3(NN / 16, BSZ / 8), 32, 0, stream>>>(h1f, h1z, s2vec, t2vec,
                                                    W2CT_h, cvec, p_x, sumx);

  // --- out = z_x @ sum_x ---
  k_final<<<BSZ, 96, 0, stream>>>(zbuf, sumx, out);
}
